// AttentionInteractionNetwork_23613730194128
// MI455X (gfx1250) — matrix operand order repaired and checked by reading, not compiled, not run on hardware
//
#include <hip/hip_runtime.h>
#include <hip/hip_bf16.h>
#include <stdint.h>

// Problem constants (from reference)
#define GN 10000
#define GE 160000
#define GD 256
#define GH 512

typedef unsigned short u16;
typedef __attribute__((ext_vector_type(16))) __bf16 v16bf;
typedef __attribute__((ext_vector_type(8)))  float  v8f;
typedef __attribute__((ext_vector_type(4)))  int    vi4;

__device__ __forceinline__ u16 f2bf(float f) {
  unsigned int u = __float_as_uint(f);
  u += 0x7FFFu + ((u >> 16) & 1u);   // round-to-nearest-even
  return (u16)(u >> 16);
}
__device__ __forceinline__ float silu_f(float x) { return x / (1.f + __expf(-x)); }

// ---------------------------------------------------------------------------
// 16-byte staging: prefer gfx1250 async global->LDS DMA (ASYNCcnt) when the
// toolchain declares the builtin; otherwise classic load+ds_store.
// Builtin signature (from clang diagnostic): (v4i AS1*, v4i AS3*, imm, imm).
// ---------------------------------------------------------------------------
#if __has_builtin(__builtin_amdgcn_global_load_async_to_lds_b128)
#define AIN_ASYNC_LDS 1
#else
#define AIN_ASYNC_LDS 0
#endif

__device__ __forceinline__ void stage_b128(const u16* __restrict__ g, u16* l) {
#if AIN_ASYNC_LDS
  __builtin_amdgcn_global_load_async_to_lds_b128(
      (__attribute__((address_space(1))) vi4*)g,
      (__attribute__((address_space(3))) vi4*)l, 0, 0);
#else
  *(uint4*)l = *(const uint4*)g;
#endif
}

__device__ __forceinline__ void stage_fence() {
#if AIN_ASYNC_LDS
#if __has_builtin(__builtin_amdgcn_s_wait_asynccnt)
  __builtin_amdgcn_s_wait_asynccnt(0);
#else
  asm volatile("s_wait_asynccnt 0" ::: "memory");
#endif
#endif
}

// ---------------------------------------------------------------------------
// WMMA bf16 GEMM: C[M,Nout] = act(A[M,K] @ W[K,Nout] + bias)
// A: bf16 row-major [M,K]; Wt: bf16 transposed weights [Nout,K].
// Block tile 128x128, K-step 32, 8 waves (4 in M x 2 in N),
// wave tile 32x64 = 2x4 fragments of 16x16 -> 8 v_wmma per K-step per wave.
// ---------------------------------------------------------------------------
#define BM 128
#define BN 128
#define BK 32
#define AP 40   // LDS row pitch (halfwords), padded vs 32 to spread banks
#define BP 40

__global__ __launch_bounds__(256) void ain_gemm_bf16(
    const u16* __restrict__ A, const u16* __restrict__ Wt,
    const float* __restrict__ bias, int M, int K, int Nout, int act,
    u16* __restrict__ Ob16, float* __restrict__ Of32)
{
  __shared__ u16 As[BM * AP];
  __shared__ u16 Bs[BN * BP];
  const int tid  = threadIdx.x;
  const int lane = tid & 31;
  const int wave = tid >> 5;
  const int wm = wave >> 1;          // 0..3 (M direction, 32 rows each)
  const int wn = wave & 1;           // 0..1 (N direction, 64 cols each)
  const int ntn = Nout / BN;
  const int m0 = (blockIdx.x / ntn) * BM;
  const int n0 = (blockIdx.x % ntn) * BN;

  v8f acc[2][4];
#pragma unroll
  for (int i = 0; i < 2; ++i)
#pragma unroll
    for (int j = 0; j < 4; ++j)
#pragma unroll
      for (int v = 0; v < 8; ++v) acc[i][j][v] = 0.f;

  union Frag { v16bf v; uint4 q[2]; };

  for (int kk = 0; kk < K; kk += BK) {
    // ---- stage A tile (BM x BK) and B tile (BN x BK): 2 chunks of 8 each
#pragma unroll
    for (int c0 = 0; c0 < 2; ++c0) {
      int c  = tid + 256 * c0;
      int rr = c >> 2, q = c & 3;
      int rg = m0 + rr; if (rg > M - 1) rg = M - 1;        // clamp; stores masked later
      stage_b128(A  + (size_t)rg * K        + kk + q * 8, &As[rr * AP + q * 8]);
      stage_b128(Wt + (size_t)(n0 + rr) * K + kk + q * 8, &Bs[rr * BP + q * 8]);
    }
    if (kk + BK < K) {
      // near-cache prefetch of next A tile (global_prefetch_b8)
      __builtin_prefetch(A + (size_t)(m0 + (tid >> 1)) * K + kk + BK, 0, 3);
    }
    stage_fence();
    __syncthreads();

    // ---- fragments per ISA 7.12.2 (wave32, 16-bit A 16x32 / B 32x16)
    const int khalf = (lane >> 4) * 8;    // A: K groups {0..7,16..23} or {8..15,24..31}
    const int kbase = (lane >> 4) * 8;    // B: K rows {0..7,16..23} or {8..15,24..31}
    Frag a[2], b[4];
#pragma unroll
    for (int i = 0; i < 2; ++i) {
      int m = wm * 32 + 16 * i + (lane & 15);
      a[i].q[0] = *(const uint4*)(&As[m * AP + khalf]);
      a[i].q[1] = *(const uint4*)(&As[m * AP + khalf + 16]);
    }
#pragma unroll
    for (int j = 0; j < 4; ++j) {
      int n = wn * 64 + 16 * j + (lane & 15);
      b[j].q[0] = *(const uint4*)(&Bs[n * BP + kbase]);
      b[j].q[1] = *(const uint4*)(&Bs[n * BP + kbase + 16]);
    }
#pragma unroll
    for (int i = 0; i < 2; ++i)
#pragma unroll
      for (int j = 0; j < 4; ++j)
        acc[i][j] = __builtin_amdgcn_wmma_f32_16x16x32_bf16(
            false, a[i].v, false, b[j].v, (short)0, acc[i][j], false, false);
    __syncthreads();
  }

  // ---- epilogue: bias + optional SiLU, store bf16 or fp32
#pragma unroll
  for (int j = 0; j < 4; ++j) {
    int col = n0 + wn * 64 + 16 * j + (lane & 15);
    float bv = bias[col];
#pragma unroll
    for (int i = 0; i < 2; ++i) {
#pragma unroll
      for (int v = 0; v < 8; ++v) {
        int row = m0 + wm * 32 + 16 * i + v + 8 * (lane >> 4);   // C layout 7.12.2
        if (row < M) {
          float x = acc[i][j][v] + bv;
          if (act) x = silu_f(x);
          if (Ob16) Ob16[(size_t)row * Nout + col] = f2bf(x);
          else      Of32[(size_t)row * Nout + col] = x;
        }
      }
    }
  }
}

// ---------------------------------------------------------------------------
// Weight transpose + fp32 -> bf16:  Wt[n*K + k] = bf16(W[k*Nout + n])
// ---------------------------------------------------------------------------
__global__ __launch_bounds__(256) void ain_wconv(const float* __restrict__ w,
                                                 u16* __restrict__ wt, int K, int Nout) {
  int idx = blockIdx.x * 256 + threadIdx.x;
  if (idx >= K * Nout) return;
  int k = idx / Nout, n = idx - k * Nout;
  wt[(size_t)n * K + k] = f2bf(w[idx]);
}

// ---------------------------------------------------------------------------
// edge_in[e] = bf16(concat(edges[e], nodes[senders[e]], nodes[receivers[e]]))
// ---------------------------------------------------------------------------
__global__ __launch_bounds__(256) void ain_edge_in(const float* __restrict__ edges,
                                                   const float* __restrict__ nodes,
                                                   const int* __restrict__ senders,
                                                   const int* __restrict__ receivers,
                                                   u16* __restrict__ edge_in) {
  int e = blockIdx.x, t = threadIdx.x;
  size_t base = (size_t)e * (3 * GD);
  int s = senders[e], rc = receivers[e];
  edge_in[base + t]            = f2bf(edges[(size_t)e  * GD + t]);
  edge_in[base + GD + t]       = f2bf(nodes[(size_t)s  * GD + t]);
  edge_in[base + 2 * GD + t]   = f2bf(nodes[(size_t)rc * GD + t]);
}

// node_in[i] = bf16(concat(nodes[i], recv_agg[i], sent_agg[i]))
__global__ __launch_bounds__(256) void ain_node_in(const float* __restrict__ nodes,
                                                   const float* __restrict__ recv_agg,
                                                   const float* __restrict__ sent_agg,
                                                   u16* __restrict__ node_in) {
  int i = blockIdx.x, t = threadIdx.x;
  size_t base = (size_t)i * (3 * GD);
  node_in[base + t]          = f2bf(nodes[(size_t)i * GD + t]);
  node_in[base + GD + t]     = f2bf(recv_agg[(size_t)i * GD + t]);
  node_in[base + 2 * GD + t] = f2bf(sent_agg[(size_t)i * GD + t]);
}

// ---------------------------------------------------------------------------
// Attention gates + polynomial cutoff. One wave32 per edge; 8 edges/block.
// ---------------------------------------------------------------------------
__global__ __launch_bounds__(256) void ain_attn(const float* __restrict__ edges,
                                                const float* __restrict__ r,
                                                const float* __restrict__ rw, const float* __restrict__ rb,
                                                const float* __restrict__ sw, const float* __restrict__ sb,
                                                float* __restrict__ attn_r, float* __restrict__ attn_s) {
  int lane = threadIdx.x & 31;
  int e = blockIdx.x * 8 + (threadIdx.x >> 5);
  if (e >= GE) return;
  const float* row = edges + (size_t)e * GD;
  float dr = 0.f, ds = 0.f;
#pragma unroll
  for (int j = 0; j < 8; ++j) {
    int d = lane + 32 * j;
    float v = row[d];
    dr += v * rw[d];
    ds += v * sw[d];
  }
#pragma unroll
  for (int o = 16; o > 0; o >>= 1) {       // wave32 reduction
    dr += __shfl_xor(dr, o, 32);
    ds += __shfl_xor(ds, o, 32);
  }
  if (lane == 0) {
    float rr = r[e];
    float x  = rr * (1.f / 6.f);
    float x4 = x * x; x4 = x4 * x4;
    float env = 1.f - 15.f * x4 + 24.f * x4 * x - 10.f * x4 * x * x;   // P=4
    float cut = (rr < 6.f) ? env : 0.f;
    attn_r[e] = cut / (1.f + __expf(-(dr + rb[0])));
    attn_s[e] = cut / (1.f + __expf(-(ds + sb[0])));
  }
}

// ---------------------------------------------------------------------------
// LayerNorm over D=256 (wave per row) with optional residual add.
// ---------------------------------------------------------------------------
__global__ __launch_bounds__(256) void ain_ln(const float* __restrict__ H,
                                              const float* __restrict__ g,
                                              const float* __restrict__ b,
                                              const float* __restrict__ residual,
                                              float* __restrict__ out, int rows) {
  int lane = threadIdx.x & 31;
  int row = blockIdx.x * 8 + (threadIdx.x >> 5);
  if (row >= rows) return;
  const float* h = H + (size_t)row * GD;
  float v[8], s = 0.f, s2 = 0.f;
#pragma unroll
  for (int j = 0; j < 8; ++j) {
    v[j] = h[lane + 32 * j];
    s += v[j]; s2 += v[j] * v[j];
  }
#pragma unroll
  for (int o = 16; o > 0; o >>= 1) {
    s  += __shfl_xor(s, o, 32);
    s2 += __shfl_xor(s2, o, 32);
  }
  float mean = s * (1.f / GD);
  float var  = s2 * (1.f / GD) - mean * mean;
  float inv  = rsqrtf(var + 1e-5f);
#pragma unroll
  for (int j = 0; j < 8; ++j) {
    int d = lane + 32 * j;
    float o = (v[j] - mean) * inv * g[d] + b[d];
    if (residual) o += residual[(size_t)row * GD + d];
    out[(size_t)row * GD + d] = o;
  }
}

// Zero fill
__global__ __launch_bounds__(256) void ain_zero(float* __restrict__ p, int n) {
  int idx = blockIdx.x * 256 + threadIdx.x;
  if (idx < n) p[idx] = 0.f;
}

// Attention-weighted segment sums (block per edge, thread per feature)
__global__ __launch_bounds__(256) void ain_segsum(const float* __restrict__ upd,
                                                  const float* __restrict__ attn_s,
                                                  const float* __restrict__ attn_r,
                                                  const int* __restrict__ senders,
                                                  const int* __restrict__ receivers,
                                                  float* __restrict__ sent_agg,
                                                  float* __restrict__ recv_agg) {
  int e = blockIdx.x, d = threadIdx.x;
  float v = upd[(size_t)e * GD + d];
  int s = senders[e], rc = receivers[e];
  atomicAdd(&sent_agg[(size_t)s  * GD + d], v * attn_s[e]);
  atomicAdd(&recv_agg[(size_t)rc * GD + d], v * attn_r[e]);
}

// out_edges += edges (residual, in place after segment sums consumed)
__global__ __launch_bounds__(256) void ain_add(float* __restrict__ out,
                                               const float* __restrict__ add, size_t n) {
  size_t idx = (size_t)blockIdx.x * 256 + threadIdx.x;
  if (idx < n) out[idx] += add[idx];
}

// ---------------------------------------------------------------------------
extern "C" void kernel_launch(void* const* d_in, const int* in_sizes, int n_in,
                              void* d_out, int out_size, void* d_ws, size_t ws_size,
                              hipStream_t stream) {
  const float* nodes     = (const float*)d_in[0];
  const float* edges     = (const float*)d_in[1];
  const float* r         = (const float*)d_in[2];
  const int*   senders   = (const int*)d_in[3];
  const int*   receivers = (const int*)d_in[4];
  const float* ew0 = (const float*)d_in[5];  const float* eb0 = (const float*)d_in[6];
  const float* ew1 = (const float*)d_in[7];  const float* eb1 = (const float*)d_in[8];
  const float* ew2 = (const float*)d_in[9];  const float* eb2 = (const float*)d_in[10];
  const float* elng = (const float*)d_in[11]; const float* elnb = (const float*)d_in[12];
  const float* nw0 = (const float*)d_in[13]; const float* nb0 = (const float*)d_in[14];
  const float* nw1 = (const float*)d_in[15]; const float* nb1 = (const float*)d_in[16];
  const float* nw2 = (const float*)d_in[17]; const float* nb2 = (const float*)d_in[18];
  const float* nlng = (const float*)d_in[19]; const float* nlnb = (const float*)d_in[20];
  const float* rw = (const float*)d_in[21];  const float* rb = (const float*)d_in[22];
  const float* sw = (const float*)d_in[23];  const float* sb = (const float*)d_in[24];

  float* out_nodes = (float*)d_out;
  float* out_edges = out_nodes + (size_t)GN * GD;

  // Workspace carve-up
  uint8_t* ws = (uint8_t*)d_ws;
  size_t off = 0;
  auto alloc = [&](size_t bytes) -> void* {
    void* p = ws + off;
    off += (bytes + 255) & ~(size_t)255;
    return p;
  };
  u16* wt_e0 = (u16*)alloc((size_t)3 * GD * GH * 2);   // [512,768]
  u16* wt_e1 = (u16*)alloc((size_t)GH * GH * 2);       // [512,512]
  u16* wt_e2 = (u16*)alloc((size_t)GD * GH * 2);       // [256,512]
  u16* wt_n0 = (u16*)alloc((size_t)3 * GD * GH * 2);
  u16* wt_n1 = (u16*)alloc((size_t)GH * GH * 2);
  u16* wt_n2 = (u16*)alloc((size_t)GD * GH * 2);
  float* attn_r = (float*)alloc((size_t)GE * 4);
  float* attn_s = (float*)alloc((size_t)GE * 4);
  float* aggs   = (float*)alloc((size_t)2 * GN * GD * 4);  // sent | recv, contiguous
  float* sent_agg = aggs;
  float* recv_agg = aggs + (size_t)GN * GD;
  u16* edge_in = (u16*)alloc((size_t)GE * 3 * GD * 2);     // also reused as eh2
  u16* eh1     = (u16*)alloc((size_t)GE * GH * 2);
  u16* eh2     = edge_in;                                  // alias: edge_in dead after layer0
  u16* node_in = (u16*)alloc((size_t)GN * 3 * GD * 2);     // also reused as nh2
  u16* nh1     = (u16*)alloc((size_t)GN * GH * 2);
  u16* nh2     = node_in;
  float* node_h = (float*)alloc((size_t)GN * GD * 4);

  // 1) weights -> bf16, transposed [Nout][K]
  ain_wconv<<<(3 * GD * GH + 255) / 256, 256, 0, stream>>>(ew0, wt_e0, 3 * GD, GH);
  ain_wconv<<<(GH * GH + 255) / 256, 256, 0, stream>>>(ew1, wt_e1, GH, GH);
  ain_wconv<<<(GH * GD + 255) / 256, 256, 0, stream>>>(ew2, wt_e2, GH, GD);
  ain_wconv<<<(3 * GD * GH + 255) / 256, 256, 0, stream>>>(nw0, wt_n0, 3 * GD, GH);
  ain_wconv<<<(GH * GH + 255) / 256, 256, 0, stream>>>(nw1, wt_n1, GH, GH);
  ain_wconv<<<(GH * GD + 255) / 256, 256, 0, stream>>>(nw2, wt_n2, GH, GD);

  // 2) gather + concat + bf16 edge inputs; 3) attention gates
  ain_edge_in<<<GE, 256, 0, stream>>>(edges, nodes, senders, receivers, edge_in);
  ain_attn<<<(GE + 7) / 8, 256, 0, stream>>>(edges, r, rw, rb, sw, sb, attn_r, attn_s);

  // 4-6) edge MLP (WMMA); layer 2 writes pre-LN fp32 straight into d_out edge region
  ain_gemm_bf16<<<(GE / BM) * (GH / BN), 256, 0, stream>>>(edge_in, wt_e0, eb0, GE, 3 * GD, GH, 1, eh1, nullptr);
  ain_gemm_bf16<<<(GE / BM) * (GH / BN), 256, 0, stream>>>(eh1, wt_e1, eb1, GE, GH, GH, 1, eh2, nullptr);
  ain_gemm_bf16<<<(GE / BM) * (GD / BN), 256, 0, stream>>>(eh2, wt_e2, eb2, GE, GH, GD, 0, nullptr, out_edges);

  // 7) LayerNorm in place -> updated_edges lives in d_out edge region
  ain_ln<<<(GE + 7) / 8, 256, 0, stream>>>(out_edges, elng, elnb, nullptr, out_edges, GE);

  // 8) zero aggs, 9) attention-weighted segment sums
  ain_zero<<<(2 * GN * GD + 255) / 256, 256, 0, stream>>>(aggs, 2 * GN * GD);
  ain_segsum<<<GE, 256, 0, stream>>>(out_edges, attn_s, attn_r, senders, receivers, sent_agg, recv_agg);

  // 10) node inputs
  ain_node_in<<<GN, 256, 0, stream>>>(nodes, recv_agg, sent_agg, node_in);

  // 11-13) node MLP (WMMA)
  int nmt = (GN + BM - 1) / BM;
  ain_gemm_bf16<<<nmt * (GH / BN), 256, 0, stream>>>(node_in, wt_n0, nb0, GN, 3 * GD, GH, 1, nh1, nullptr);
  ain_gemm_bf16<<<nmt * (GH / BN), 256, 0, stream>>>(nh1, wt_n1, nb1, GN, GH, GH, 1, nh2, nullptr);
  ain_gemm_bf16<<<nmt * (GD / BN), 256, 0, stream>>>(nh2, wt_n2, nb2, GN, GH, GD, 0, nullptr, node_h);

  // 14) node LN + residual -> d_out node region
  ain_ln<<<(GN + 7) / 8, 256, 0, stream>>>(node_h, nlng, nlnb, nodes, out_nodes, GN);

  // 15) edge residual (after segment sums have consumed updated_edges)
  ain_add<<<(int)(((size_t)GE * GD + 255) / 256), 256, 0, stream>>>(out_edges, edges, (size_t)GE * GD);
}
